// GATNet_20968030339560
// MI455X (gfx1250) — matrix operand order repaired and checked by reading, not compiled, not run on hardware
//
#include <hip/hip_runtime.h>
#include <hip/hip_bf16.h>
#include <math.h>

#define NN      50000
#define EE      800000
#define HIDD    128
#define NHEADS  8
#define CC      16
#define NCLS    10
#define NGR     64
#define NCONV   3
#define BN_EPS  1e-5f
#define NEG_SL  0.2f

typedef __attribute__((ext_vector_type(16))) _Float16 v16h;
typedef __attribute__((ext_vector_type(8)))  _Float16 v8h;
typedef __attribute__((ext_vector_type(8)))  float    v8f;

// ---- ordered-float <-> uint monotone mapping (for atomicMax on f32) ----
__device__ __forceinline__ unsigned f2o(float f) {
    unsigned u = __float_as_uint(f);
    return (u & 0x80000000u) ? ~u : (u | 0x80000000u);
}
__device__ __forceinline__ float o2f(unsigned o) {
    unsigned u = (o & 0x80000000u) ? (o ^ 0x80000000u) : ~o;
    return __uint_as_float(u);
}

// ------------------------- fill kernels -------------------------
__global__ void fill_f32(float* __restrict__ p, float v, int n) {
    int i = blockIdx.x * blockDim.x + threadIdx.x;
    if (i < n) p[i] = v;
}
__global__ void fill_neginf_ord(unsigned* __restrict__ p, int n) {
    int i = blockIdx.x * blockDim.x + threadIdx.x;
    if (i < n) p[i] = f2o(-1.0e30f);
}

// ---------------- per-layer weight prep: fold BN, f16-swizzle ----------------
// bn(x) = a*x + c with a = g*rsqrt(v+eps), c = b - m*a
// bn(x)@W = x @ (diag(a)W)  +  c@W
// Writes Whf in WMMA B-fragment order: frag (kt,nt), lane l, elem i ->
//   n = nt*16 + (l&15), k = kt*32 + (l>=16 ? 8 : 0) + (i&7) + (i>=8 ? 16 : 0)   (16-bit B 32x16 layout)
__global__ __launch_bounds__(256) void prep_weights(
        const float* __restrict__ bnp, const float* __restrict__ W,
        const float* __restrict__ extra_bias,
        _Float16* __restrict__ Whf, float* __restrict__ rowbias) {
    __shared__ float sa[HIDD], sc[HIDD];
    int t = threadIdx.x;
    if (t < HIDD) {
        float g = bnp[t], b = bnp[HIDD + t], m = bnp[2 * HIDD + t], v = bnp[3 * HIDD + t];
        float a = g * rsqrtf(v + BN_EPS);
        sa[t] = a;
        sc[t] = b - m * a;
    }
    __syncthreads();
    if (t < HIDD) {
        float s = extra_bias ? extra_bias[t] : 0.0f;
        for (int k = 0; k < HIDD; ++k) s += sc[k] * W[k * HIDD + t];
        rowbias[t] = s;
    }
    for (int idx = t; idx < HIDD * HIDD; idx += 256) {
        int i    = idx & 15;
        int lane = (idx >> 4) & 31;
        int tile = idx >> 9;          // (kt*8 + nt), kt in 0..3, nt in 0..7
        int kt = tile >> 3, nt = tile & 7;
        int n = nt * 16 + (lane & 15);
        int k = kt * 32 + ((lane >= 16) ? 8 : 0) + (i & 7) + ((i & 8) ? 16 : 0);
        Whf[idx] = (_Float16)(sa[k] * W[k * HIDD + n]);
    }
}

// ---------------- WMMA GEMM: Y[nrows,128] = X[nrows,128] @ Wfolded + rowbias ----------------
// 8 waves/block, each wave: 16 rows x 128 cols = 8 accumulators, K-loop 4 x K32.
__global__ __launch_bounds__(256) void gemm128_wmma(
        const float* __restrict__ X, const _Float16* __restrict__ Whf,
        const float* __restrict__ rowbias, float* __restrict__ Y,
        int nrows, int do_relu) {
    int wave = threadIdx.x >> 5;
    int lane = threadIdx.x & 31;
    int rowbase = blockIdx.x * 128 + wave * 16;
    int arow = rowbase + (lane & 15);

    v8f acc[8];
#pragma unroll
    for (int nt = 0; nt < 8; ++nt)
#pragma unroll
        for (int r = 0; r < 8; ++r) acc[nt][r] = 0.0f;

#pragma unroll
    for (int kt = 0; kt < 4; ++kt) {
        // A fragment: 16-bit A 16x32 layout. lane<16: K {0..7,16..23}; lane>=16: +8.
        union { v16h v; v8h h[2]; _Float16 e[16]; } A;
        if (arow < nrows) {
            int kbase = kt * 32 + ((lane >= 16) ? 8 : 0);
            const float4* xp = (const float4*)(X + (size_t)arow * HIDD + kbase);
            float4 x0 = xp[0], x1 = xp[1];
            const float4* xq = (const float4*)(X + (size_t)arow * HIDD + kbase + 16);
            float4 x2 = xq[0], x3 = xq[1];
            A.e[0]  = (_Float16)x0.x; A.e[1]  = (_Float16)x0.y;
            A.e[2]  = (_Float16)x0.z; A.e[3]  = (_Float16)x0.w;
            A.e[4]  = (_Float16)x1.x; A.e[5]  = (_Float16)x1.y;
            A.e[6]  = (_Float16)x1.z; A.e[7]  = (_Float16)x1.w;
            A.e[8]  = (_Float16)x2.x; A.e[9]  = (_Float16)x2.y;
            A.e[10] = (_Float16)x2.z; A.e[11] = (_Float16)x2.w;
            A.e[12] = (_Float16)x3.x; A.e[13] = (_Float16)x3.y;
            A.e[14] = (_Float16)x3.z; A.e[15] = (_Float16)x3.w;
        } else {
#pragma unroll
            for (int q = 0; q < 16; ++q) A.e[q] = (_Float16)0.0f;
        }
#pragma unroll
        for (int nt = 0; nt < 8; ++nt) {
            union { v16h v; v8h h[2]; } B;
            const v8h* bp = (const v8h*)(Whf + ((size_t)((kt * 8 + nt) * 32 + lane)) * 16);
            B.h[0] = bp[0];
            B.h[1] = bp[1];
            acc[nt] = __builtin_amdgcn_wmma_f32_16x16x32_f16(
                false, A.v, false, B.v, (short)0, acc[nt], false, false);
        }
    }

    // Epilogue. C/D layout: VGPR r, lanes 0-15 -> M=r, lanes 16-31 -> M=8+r; N = lane&15.
    int mbase = rowbase + ((lane >= 16) ? 8 : 0);
    int col0 = lane & 15;
#pragma unroll
    for (int nt = 0; nt < 8; ++nt) {
        int col = nt * 16 + col0;
        float rb = rowbias[col];
#pragma unroll
        for (int r = 0; r < 8; ++r) {
            int row = mbase + r;
            if (row < nrows) {
                float v = acc[nt][r] + rb;
                if (do_relu) v = fmaxf(v, 0.0f);
                Y[(size_t)row * HIDD + col] = v;
            }
        }
    }
}

// ---------------- attention coefficients per node/head ----------------
__global__ void alpha_kernel(const float* __restrict__ hmat,
                             const float* __restrict__ asrc, const float* __restrict__ adst,
                             float* __restrict__ out_s, float* __restrict__ out_d) {
    int gid = blockIdx.x * blockDim.x + threadIdx.x;
    if (gid >= NN * NHEADS) return;
    int n = gid >> 3, hh = gid & 7;
    const float* hp = hmat + (size_t)n * HIDD + hh * CC;
    const float* ap = asrc + hh * CC;
    const float* dp = adst + hh * CC;
    float ss = 0.0f, sd = 0.0f;
#pragma unroll
    for (int c = 0; c < CC; ++c) {
        float hv = hp[c];
        ss += hv * ap[c];
        sd += hv * dp[c];
    }
    out_s[gid] = ss;
    out_d[gid] = sd;
}

__device__ __forceinline__ void edge_sd(int e, const int* src, const int* dst, int& s, int& d) {
    if (e < EE) { s = src[e]; d = dst[e]; }
    else        { s = e - EE; d = e - EE; }   // self loops
}

// ---------------- segment max over dst (ordered-uint atomicMax) ----------------
__global__ void edge_max_kernel(const int* __restrict__ src, const int* __restrict__ dst,
                                const float* __restrict__ as_, const float* __restrict__ ad_,
                                unsigned* __restrict__ mx) {
    int gid = blockIdx.x * blockDim.x + threadIdx.x;
    if (gid >= (EE + NN) * NHEADS) return;
    int e = gid >> 3, hh = gid & 7;
    int s, d; edge_sd(e, src, dst, s, d);
    float lg = as_[s * NHEADS + hh] + ad_[d * NHEADS + hh];
    lg = (lg > 0.0f) ? lg : NEG_SL * lg;
    atomicMax(&mx[d * NHEADS + hh], f2o(lg));
}

// ---------------- exp(logit - max), accumulate denom ----------------
__global__ void edge_exp_kernel(const int* __restrict__ src, const int* __restrict__ dst,
                                const float* __restrict__ as_, const float* __restrict__ ad_,
                                const unsigned* __restrict__ mx,
                                float* __restrict__ ex, float* __restrict__ denom) {
    int gid = blockIdx.x * blockDim.x + threadIdx.x;
    if (gid >= (EE + NN) * NHEADS) return;
    int e = gid >> 3, hh = gid & 7;
    int s, d; edge_sd(e, src, dst, s, d);
    float lg = as_[s * NHEADS + hh] + ad_[d * NHEADS + hh];
    lg = (lg > 0.0f) ? lg : NEG_SL * lg;
    float ev = expf(lg - o2f(mx[d * NHEADS + hh]));
    ex[gid] = ev;
    atomicAdd(&denom[d * NHEADS + hh], ev);
}

// ---------------- weighted message scatter-add ----------------
__global__ void edge_agg_kernel(const int* __restrict__ src, const int* __restrict__ dst,
                                const float* __restrict__ ex, const float* __restrict__ denom,
                                const float* __restrict__ hmat, float* __restrict__ out) {
    int gid = blockIdx.x * blockDim.x + threadIdx.x;
    if (gid >= (EE + NN) * HIDD) return;
    int e = gid >> 7, j = gid & 127, hh = j >> 4;
    int s, d; edge_sd(e, src, dst, s, d);
    float coef = ex[e * NHEADS + hh] / (denom[d * NHEADS + hh] + 1e-16f);
    atomicAdd(&out[(size_t)d * HIDD + j], coef * hmat[(size_t)s * HIDD + j]);
}

__global__ void bias_relu_kernel(float* __restrict__ h, const float* __restrict__ b) {
    int gid = blockIdx.x * blockDim.x + threadIdx.x;
    if (gid >= NN * HIDD) return;
    int j = gid & 127;
    h[gid] = fmaxf(h[gid] + b[j], 0.0f);
}

__global__ void pool_kernel(const float* __restrict__ h, const int* __restrict__ batch,
                            float* __restrict__ g) {
    int gid = blockIdx.x * blockDim.x + threadIdx.x;
    if (gid >= NN * HIDD) return;
    int n = gid >> 7, j = gid & 127;
    atomicAdd(&g[batch[n] * HIDD + j], h[gid]);
}

// ---------------- tiny head: fc + BN + classifier + log_softmax ----------------
__global__ __launch_bounds__(128) void head_kernel(
        const float* __restrict__ gpool, const float* __restrict__ bnfc,
        const float* __restrict__ fcw, const float* __restrict__ fcb,
        const float* __restrict__ bnh, const float* __restrict__ wcls,
        const float* __restrict__ bcls, float* __restrict__ out) {
    __shared__ float t[HIDD];
    __shared__ float t2[HIDD];
    __shared__ float lg[NCLS];
    int j = threadIdx.x;
    for (int g = 0; g < NGR; ++g) {
        float a = bnfc[j] * rsqrtf(bnfc[3 * HIDD + j] + BN_EPS);
        t[j] = a * (gpool[g * HIDD + j] - bnfc[2 * HIDD + j]) + bnfc[HIDD + j];
        __syncthreads();
        float s = fcb[j];
        for (int k = 0; k < HIDD; ++k) s += t[k] * fcw[k * HIDD + j];
        s = fmaxf(s, 0.0f);
        float ah = bnh[j] * rsqrtf(bnh[3 * HIDD + j] + BN_EPS);
        t2[j] = ah * (s - bnh[2 * HIDD + j]) + bnh[HIDD + j];
        __syncthreads();
        if (j < NCLS) {
            float l = bcls[j];
            for (int k = 0; k < HIDD; ++k) l += t2[k] * wcls[k * NCLS + j];
            lg[j] = l;
        }
        __syncthreads();
        if (j == 0) {
            float m = lg[0];
            for (int i = 1; i < NCLS; ++i) m = fmaxf(m, lg[i]);
            float s2 = 0.0f;
            for (int i = 0; i < NCLS; ++i) s2 += expf(lg[i] - m);
            float ls = m + logf(s2);
            for (int i = 0; i < NCLS; ++i) out[g * NCLS + i] = lg[i] - ls;
        }
        __syncthreads();
    }
}

// ------------------------------- host launcher -------------------------------
extern "C" void kernel_launch(void* const* d_in, const int* in_sizes, int n_in,
                              void* d_out, int out_size, void* d_ws, size_t ws_size,
                              hipStream_t stream) {
    const float* x        = (const float*)d_in[0];
    const int*   ei       = (const int*)d_in[1];
    const int*   src      = ei;
    const int*   dst      = ei + EE;
    const int*   batch    = (const int*)d_in[2];
    const float* bn_feat  = (const float*)d_in[3];
    const float* w_feat   = (const float*)d_in[4];
    const float* b_feat   = (const float*)d_in[5];
    const float* bns_conv = (const float*)d_in[6];
    const float* gat_w    = (const float*)d_in[7];
    const float* gat_as   = (const float*)d_in[8];
    const float* gat_ad   = (const float*)d_in[9];
    const float* gat_b    = (const float*)d_in[10];
    const float* bns_fc   = (const float*)d_in[11];
    const float* fc_w     = (const float*)d_in[12];
    const float* fc_b     = (const float*)d_in[13];
    const float* bn_hid   = (const float*)d_in[14];
    const float* w_class  = (const float*)d_in[15];
    const float* b_class  = (const float*)d_in[16];
    float* out = (float*)d_out;

    char* ws = (char*)d_ws;
    size_t off = 0;
    auto carve = [&](size_t bytes) -> void* {
        void* p = (void*)(ws + off);
        off += (bytes + 255) & ~(size_t)255;
        return p;
    };
    float*     bufA    = (float*)carve((size_t)NN * HIDD * 4);
    float*     bufB    = (float*)carve((size_t)NN * HIDD * 4);   // hmat (per-layer GEMM out)
    float*     bufC    = (float*)carve((size_t)NN * HIDD * 4);
    float*     as_     = (float*)carve((size_t)NN * NHEADS * 4);
    float*     ad_     = (float*)carve((size_t)NN * NHEADS * 4);
    unsigned*  mx      = (unsigned*)carve((size_t)NN * NHEADS * 4);
    float*     denom   = (float*)carve((size_t)NN * NHEADS * 4);
    float*     ex      = (float*)carve((size_t)(EE + NN) * NHEADS * 4);
    float*     gpool   = (float*)carve((size_t)NGR * HIDD * 4);
    _Float16*  Whf     = (_Float16*)carve((size_t)HIDD * HIDD * 2);
    float*     rowbias = (float*)carve((size_t)HIDD * 4);

    const int T = 256;
    const int gemm_blocks = (NN + 127) / 128;
    const int nh_blocks   = (NN * NHEADS + T - 1) / T;
    const int eh_blocks   = ((EE + NN) * NHEADS + T - 1) / T;
    const int ec_blocks   = (int)(((size_t)(EE + NN) * HIDD + T - 1) / T);
    const int nc_blocks   = (NN * HIDD + T - 1) / T;

    // Feature layer: relu(bn_feat(x) @ w_feat + b_feat)
    prep_weights<<<1, 256, 0, stream>>>(bn_feat, w_feat, b_feat, Whf, rowbias);
    gemm128_wmma<<<gemm_blocks, 256, 0, stream>>>(x, Whf, rowbias, bufA, NN, 1);

    float* hcur = bufA;
    float* hnext = bufC;
    for (int l = 0; l < NCONV; ++l) {
        // h = bn_l(hcur);  hmat = h @ W_l   (BN folded into GEMM)
        prep_weights<<<1, 256, 0, stream>>>(bns_conv + (size_t)l * 4 * HIDD,
                                            gat_w + (size_t)l * HIDD * HIDD,
                                            nullptr, Whf, rowbias);
        gemm128_wmma<<<gemm_blocks, 256, 0, stream>>>(hcur, Whf, rowbias, bufB, NN, 0);

        alpha_kernel<<<nh_blocks, T, 0, stream>>>(bufB,
                                                  gat_as + (size_t)l * NHEADS * CC,
                                                  gat_ad + (size_t)l * NHEADS * CC,
                                                  as_, ad_);
        fill_neginf_ord<<<nh_blocks, T, 0, stream>>>(mx, NN * NHEADS);
        fill_f32<<<nh_blocks, T, 0, stream>>>(denom, 0.0f, NN * NHEADS);
        fill_f32<<<nc_blocks, T, 0, stream>>>(hnext, 0.0f, NN * HIDD);

        edge_max_kernel<<<eh_blocks, T, 0, stream>>>(src, dst, as_, ad_, mx);
        edge_exp_kernel<<<eh_blocks, T, 0, stream>>>(src, dst, as_, ad_, mx, ex, denom);
        edge_agg_kernel<<<ec_blocks, T, 0, stream>>>(src, dst, ex, denom, bufB, hnext);
        bias_relu_kernel<<<nc_blocks, T, 0, stream>>>(hnext, gat_b + (size_t)l * HIDD);

        float* tmp = hcur; hcur = hnext; hnext = tmp;
    }

    // global_add_pool + head
    fill_f32<<<(NGR * HIDD + T - 1) / T, T, 0, stream>>>(gpool, 0.0f, NGR * HIDD);
    pool_kernel<<<nc_blocks, T, 0, stream>>>(hcur, batch, gpool);
    head_kernel<<<1, 128, 0, stream>>>(gpool, bns_fc, fc_w, fc_b, bn_hid,
                                       w_class, b_class, out);
}
